// SelfAttention_49263274885897
// MI455X (gfx1250) — matrix operand order repaired and checked by reading, not compiled, not run on hardware
//
#include <hip/hip_runtime.h>

#define B_CONST 2
#define S_CONST 2048
#define DIM_CONST 4096
#define NH 32
#define NKV 8
#define HD 128
#define NREP (NH / NKV)

typedef __attribute__((ext_vector_type(16))) __bf16 v16bf;
typedef __attribute__((ext_vector_type(8)))  float  v8f;
typedef __attribute__((ext_vector_type(4)))  unsigned int u32x4;
typedef __attribute__((ext_vector_type(8)))  int          i32x8;
typedef __attribute__((ext_vector_type(4)))  int          i32x4;

union Frag16 {
    v16bf v;
    unsigned int u[8];
    unsigned short s[16];
};

__device__ __forceinline__ unsigned short f2bf(float f) {
    union { float f; unsigned int u; } x; x.f = f;
    unsigned int r = x.u + 0x7FFFu + ((x.u >> 16) & 1u);
    return (unsigned short)(r >> 16);
}
__device__ __forceinline__ float bf2f(unsigned short b) {
    union { float f; unsigned int u; } x; x.u = ((unsigned int)b) << 16;
    return x.f;
}

// A-matrix (16x32 bf16) element K-offset for vgpr-pair v (0..7), lane-half hf
__device__ __forceinline__ int a_koff(int v, int hf) {
    int j = 2 * v;
    return (j < 8) ? (hf * 8 + j) : (16 + hf * 8 + (j - 8));
}

__device__ __forceinline__ v8f wmma_bf16(const Frag16& a, const Frag16& b, v8f c) {
    return __builtin_amdgcn_wmma_f32_16x16x32_bf16(
        false, a.v, false, b.v, (short)0, c, false, false);
}

// DPP row_ror:N (rotate within 16-lane row, pure VALU, no LDS) for reductions
template <int N>
__device__ __forceinline__ float row_ror(float x) {
    int i = __float_as_int(x);
    i = __builtin_amdgcn_update_dpp(i, i, 0x120 + N, 0xF, 0xF, false);
    return __int_as_float(i);
}
__device__ __forceinline__ float rowmax16(float x) {
    x = fmaxf(x, row_ror<8>(x));
    x = fmaxf(x, row_ror<4>(x));
    x = fmaxf(x, row_ror<2>(x));
    x = fmaxf(x, row_ror<1>(x));
    return x;
}
__device__ __forceinline__ float rowsum16(float x) {
    x += row_ror<8>(x);
    x += row_ror<4>(x);
    x += row_ror<2>(x);
    x += row_ror<1>(x);
    return x;
}

// ---------------------------------------------------------------------------
// Tensor Data Mover: 2D tile load (bf16 elements, data_size=2B) into LDS.
// D# built per CDNA5 ISA 8.3-8.6. Issued by one wave; EXEC ignored by HW.
// ---------------------------------------------------------------------------
__device__ __forceinline__ void tdm_load_2d(
    unsigned int lds_addr, const void* gptr,
    unsigned int tensor_d0, unsigned int tensor_d1,
    unsigned int tile_d0, unsigned int tile_d1,
    unsigned int stride_d0)
{
    unsigned long long ga = (unsigned long long)gptr;
    u32x4 g0;
    g0[0] = 1u;                                   // count=1, user descriptor
    g0[1] = lds_addr;                             // LDS byte address
    g0[2] = (unsigned int)(ga & 0xFFFFFFFFu);     // global addr [31:0]
    g0[3] = (unsigned int)((ga >> 32) & 0x01FFFFFFu) | (2u << 30); // [56:32] + type=2

    i32x8 g1;
    g1[0] = (int)(1u << 16);                                      // data_size=1 (2B)
    g1[1] = (int)((tensor_d0 & 0xFFFFu) << 16);                   // dim0[15:0]
    g1[2] = (int)(((tensor_d0 >> 16) & 0xFFFFu) | ((tensor_d1 & 0xFFFFu) << 16));
    g1[3] = (int)(((tensor_d1 >> 16) & 0xFFFFu) | ((tile_d0 & 0xFFFFu) << 16));
    g1[4] = (int)(tile_d1 & 0xFFFFu);                             // tile_dim1, tile_dim2=0
    g1[5] = (int)stride_d0;                                       // dim0_stride[31:0]
    g1[6] = 0;
    g1[7] = 0;

    i32x4 g2; g2[0] = 1; g2[1] = 1; g2[2] = 0; g2[3] = 0;         // dim2=1, dim3=1
    i32x4 g3; g3[0] = 0; g3[1] = (int)(1u << 16); g3[2] = 0; g3[3] = 0; // dim4=1

#if defined(__clang_major__) && (__clang_major__ >= 23)
    i32x8 z8 = {0, 0, 0, 0, 0, 0, 0, 0};
    __builtin_amdgcn_tensor_load_to_lds(g0, g1, g2, g3, z8, 0);
#else
    __builtin_amdgcn_tensor_load_to_lds(g0, g1, g2, g3, 0);
#endif
}

__device__ __forceinline__ unsigned int lds_addr_of(const void* p) {
    return (unsigned int)(unsigned long long)p;   // low 32 bits = LDS byte offset
}

// ---------------------------------------------------------------------------
// C[M,N] = A[M,K] * Wt[N,K]^T : bf16 inputs (Wt pre-transposed), f32 acc via
// WMMA, bf16 or f32 out. 8 waves, tile 128x128, k-step 32, TDM double-buffered.
// ---------------------------------------------------------------------------
template <bool BF16OUT>
__global__ __launch_bounds__(256) void gemm_tdm(
    const unsigned short* __restrict__ A,
    const unsigned short* __restrict__ Wt,
    void* __restrict__ C, int M, int N, int K)
{
    __shared__ __align__(16) unsigned short As[2][128][32];
    __shared__ __align__(16) unsigned short Bs[2][128][32];

    const int tid  = threadIdx.x;
    const int wave = tid >> 5, lane = tid & 31;
    const int hf   = lane >> 4, ln = lane & 15;
    const int wm   = wave >> 1, wn = wave & 1;
    const int m0   = blockIdx.y * 128, n0 = blockIdx.x * 128;
    const bool issuer = (wave == 0);

    v8f acc[2][4];
#pragma unroll
    for (int i = 0; i < 2; ++i)
#pragma unroll
        for (int j = 0; j < 4; ++j)
#pragma unroll
            for (int r = 0; r < 8; ++r) acc[i][j][r] = 0.0f;

    if (issuer) {
        tdm_load_2d(lds_addr_of(&As[0][0][0]), A + (size_t)m0 * K, K, M, 32, 128, K);
        tdm_load_2d(lds_addr_of(&Bs[0][0][0]), Wt + (size_t)n0 * K, K, N, 32, 128, K);
    }

    const int nk = K / 32;
    for (int ki = 0; ki < nk; ++ki) {
        const int cur = ki & 1;
        if (issuer) __builtin_amdgcn_s_wait_tensorcnt(0);
        __syncthreads();
        if (issuer && (ki + 1) < nk) {
            const int k0n = (ki + 1) * 32;
            tdm_load_2d(lds_addr_of(&As[cur ^ 1][0][0]),
                        A + (size_t)m0 * K + k0n, K, M, 32, 128, K);
            tdm_load_2d(lds_addr_of(&Bs[cur ^ 1][0][0]),
                        Wt + (size_t)n0 * K + k0n, K, N, 32, 128, K);
        }

        Frag16 af[2], bf[4];
#pragma unroll
        for (int mf = 0; mf < 2; ++mf) {
            const unsigned short* rp = &As[cur][wm * 32 + mf * 16 + ln][0];
#pragma unroll
            for (int v = 0; v < 8; ++v)
                af[mf].u[v] = *(const unsigned int*)(rp + a_koff(v, hf));
        }
#pragma unroll
        for (int nf = 0; nf < 4; ++nf) {
            const unsigned short* rp = &Bs[cur][wn * 64 + nf * 16 + ln][0];
#pragma unroll
            for (int v = 0; v < 8; ++v)
                bf[nf].u[v] = *(const unsigned int*)(rp + a_koff(v, hf));
        }
#pragma unroll
        for (int mf = 0; mf < 2; ++mf)
#pragma unroll
            for (int nf = 0; nf < 4; ++nf)
                acc[mf][nf] = wmma_bf16(af[mf], bf[nf], acc[mf][nf]);
    }

#pragma unroll
    for (int mf = 0; mf < 2; ++mf)
#pragma unroll
        for (int nf = 0; nf < 4; ++nf)
#pragma unroll
            for (int r = 0; r < 8; ++r) {
                int row = m0 + wm * 32 + mf * 16 + hf * 8 + r;
                int col = n0 + wn * 64 + nf * 16 + ln;
                if (BF16OUT)
                    ((unsigned short*)C)[(size_t)row * N + col] = f2bf(acc[mf][nf][r]);
                else
                    ((float*)C)[(size_t)row * N + col] = acc[mf][nf][r];
            }
}

// ---------------------------------------------------------------------------
__global__ void f32_to_bf16(const float* __restrict__ src,
                            unsigned short* __restrict__ dst, int n)
{
    int t = blockIdx.x * blockDim.x + threadIdx.x;
    if (t < n) dst[t] = f2bf(src[t]);
}

// W[K,N] f32 -> Wt[N,K] bf16 (tiled transpose via LDS)
__global__ __launch_bounds__(256) void transpose_conv(
    const float* __restrict__ w, unsigned short* __restrict__ wt, int K, int N)
{
    __shared__ unsigned short tile[32][33];
    const int kb = blockIdx.y * 32, nb = blockIdx.x * 32;
    const int tx = threadIdx.x & 31, ty = threadIdx.x >> 5;
#pragma unroll
    for (int i = 0; i < 32; i += 8)
        tile[ty + i][tx] = f2bf(w[(size_t)(kb + ty + i) * N + nb + tx]);
    __syncthreads();
#pragma unroll
    for (int i = 0; i < 32; i += 8)
        wt[(size_t)(nb + ty + i) * K + kb + tx] = tile[tx][ty + i];
}

// In-place RoPE on bf16 (B,S,nheads,HD); scale folds 1/sqrt(HD) into Q.
__global__ void rope_inplace_bf16(unsigned short* __restrict__ t,
                                  const float* __restrict__ cs,
                                  const float* __restrict__ sn,
                                  int nheads, float scale, int total)
{
    int i = blockIdx.x * blockDim.x + threadIdx.x;
    if (i >= total) return;
    int p    = i & 63;
    int rest = i >> 6;
    int s    = (rest / nheads) % S_CONST;
    size_t base = (size_t)rest * HD + 2 * p;
    float c  = cs[s * (HD / 2) + p];
    float si = sn[s * (HD / 2) + p];
    float xr = bf2f(t[base]), xi = bf2f(t[base + 1]);
    t[base]     = f2bf((xr * c - xi * si) * scale);
    t[base + 1] = f2bf((xr * si + xi * c) * scale);
}

// ---------------------------------------------------------------------------
// Flash attention (non-causal). Block = 4 waves, 64 q-rows of one (b,h).
// Double-buffered staging: TDM loads K(i+1) while chunk i computes; V(i+1)
// prefetched to registers, scattered (transposed) to LDS after compute.
// DPP row-rotate reductions for online softmax (no LDS permutes).
// Output bf16 written IN PLACE over the Q buffer.
// ---------------------------------------------------------------------------
__global__ __launch_bounds__(128) void flash_attn(
    const unsigned short* __restrict__ Kb,   // bf16 (B,S,NKV,HD)
    const unsigned short* __restrict__ Vb,   // bf16 (B,S,NKV,HD)
    unsigned short* __restrict__ QO)         // bf16 (B,S,NH,HD): Q in, O out
{
    __shared__ __align__(16) unsigned short Ks[2][32][HD];   // [buf][key][hd]
    __shared__ __align__(16) unsigned short Vt[2][HD][32];   // [buf][hd][key]
    __shared__ __align__(16) unsigned short Pl[4][16][32];   // per-wave P tile

    const int tid  = threadIdx.x;
    const int wave = tid >> 5, lane = tid & 31;
    const int hf   = lane >> 4, ln = lane & 15;
    const bool issuer = (wave == 0);

    int bid = blockIdx.x;
    const int nqt = S_CONST / 64;
    const int qt  = bid % nqt;  bid /= nqt;
    const int h   = bid % NH;   bid /= NH;
    const int b   = bid;
    const int hk  = h / NREP;
    const int q0  = qt * 64 + wave * 16;

    Frag16 aq[4];
    {
        const unsigned short* qrow =
            QO + (size_t)((b * S_CONST + q0 + ln) * NH + h) * HD;
#pragma unroll
        for (int kf = 0; kf < 4; ++kf)
#pragma unroll
            for (int v = 0; v < 8; ++v)
                aq[kf].u[v] = *(const unsigned int*)(qrow + kf * 32 + a_koff(v, hf));
    }

    float mrow[8], lrow[8];
    v8f acc[8];
#pragma unroll
    for (int r = 0; r < 8; ++r) { mrow[r] = -3.0e38f; lrow[r] = 0.0f; }
#pragma unroll
    for (int n = 0; n < 8; ++n)
#pragma unroll
        for (int r = 0; r < 8; ++r) acc[n][r] = 0.0f;

    const int skey = tid >> 2, shb = (tid & 3) * 32;  // V staging: key, hd-base
    const size_t kvRow = (size_t)NKV * HD;            // elems between keys

    unsigned int vreg[8];                             // 8 bf16 pairs prefetch

    // ---- prologue: stage chunk 0 into buffer 0 ----
    if (issuer)
        tdm_load_2d(lds_addr_of(&Ks[0][0][0]),
                    Kb + (size_t)(b * S_CONST * NKV + hk) * HD,
                    HD, 32, HD, 32, NKV * HD);
    {
        const unsigned int* vs = (const unsigned int*)
            (Vb + (size_t)((b * S_CONST + skey) * NKV + hk) * HD + shb);
#pragma unroll
        for (int i = 0; i < 8; ++i) vreg[i] = vs[i];
#pragma unroll
        for (int i = 0; i < 8; ++i) {
            Vt[0][shb + 2 * i + 0][skey] = (unsigned short)(vreg[i] & 0xFFFFu);
            Vt[0][shb + 2 * i + 1][skey] = (unsigned short)(vreg[i] >> 16);
        }
    }
    if (issuer) __builtin_amdgcn_s_wait_tensorcnt(0);
    __syncthreads();

    const int nch = S_CONST / 32;
    for (int ci = 0; ci < nch; ++ci) {
        const int cur  = ci & 1;
        const bool more = (ci + 1) < nch;

        // ---- kick off staging of chunk ci+1 (TDM K + V reg prefetch) ----
        if (more) {
            const int c1 = (ci + 1) * 32;
            if (issuer)
                tdm_load_2d(lds_addr_of(&Ks[cur ^ 1][0][0]),
                            Kb + (size_t)((b * S_CONST + c1) * NKV + hk) * HD,
                            HD, 32, HD, 32, NKV * HD);
            const unsigned int* vs = (const unsigned int*)
                (Vb + (size_t)((b * S_CONST + c1 + skey) * NKV + hk) * HD + shb);
#pragma unroll
            for (int i = 0; i < 8; ++i) vreg[i] = vs[i];
        }

        // ---- S = Q*K^T : kf outer, two independent WMMA chains inner ----
        v8f s0, s1;
#pragma unroll
        for (int r = 0; r < 8; ++r) { s0[r] = 0.0f; s1[r] = 0.0f; }
#pragma unroll
        for (int kf = 0; kf < 4; ++kf) {
            Frag16 b0, b1;
            const unsigned short* rp0 = &Ks[cur][ln][kf * 32];
            const unsigned short* rp1 = &Ks[cur][16 + ln][kf * 32];
#pragma unroll
            for (int v = 0; v < 8; ++v) {
                b0.u[v] = *(const unsigned int*)(rp0 + a_koff(v, hf));
                b1.u[v] = *(const unsigned int*)(rp1 + a_koff(v, hf));
            }
            s0 = wmma_bf16(aq[kf], b0, s0);
            s1 = wmma_bf16(aq[kf], b1, s1);
        }

        // ---- online softmax; row reductions via DPP row_ror (VALU only) ----
        float p0[8], p1[8];
#pragma unroll
        for (int r = 0; r < 8; ++r) {
            float mc = rowmax16(fmaxf(s0[r], s1[r]));
            float mn = fmaxf(mrow[r], mc);
            float al = __expf(mrow[r] - mn);
            mrow[r] = mn;
            p0[r] = __expf(s0[r] - mn);
            p1[r] = __expf(s1[r] - mn);
            float ls = rowsum16(p0[r] + p1[r]);
            lrow[r] = lrow[r] * al + ls;
#pragma unroll
            for (int n = 0; n < 8; ++n) acc[n][r] *= al;
        }

        // ---- C-layout -> A-layout through per-wave LDS tile ----
#pragma unroll
        for (int r = 0; r < 8; ++r) {
            Pl[wave][hf * 8 + r][ln]      = f2bf(p0[r]);
            Pl[wave][hf * 8 + r][16 + ln] = f2bf(p1[r]);
        }
        __syncthreads();

        Frag16 pa;
#pragma unroll
        for (int v = 0; v < 8; ++v)
            pa.u[v] = *(const unsigned int*)(&Pl[wave][ln][0] + a_koff(v, hf));

#pragma unroll
        for (int n = 0; n < 8; ++n) {
            Frag16 bv;
            const unsigned short* rp = &Vt[cur][n * 16 + ln][0];
#pragma unroll
            for (int v = 0; v < 8; ++v)
                bv.u[v] = *(const unsigned int*)(rp + a_koff(v, hf));
            acc[n] = wmma_bf16(pa, bv, acc[n]);
        }

        // ---- commit staged chunk ci+1, drain TDM, flip buffers ----
        if (more) {
#pragma unroll
            for (int i = 0; i < 8; ++i) {
                Vt[cur ^ 1][shb + 2 * i + 0][skey] = (unsigned short)(vreg[i] & 0xFFFFu);
                Vt[cur ^ 1][shb + 2 * i + 1][skey] = (unsigned short)(vreg[i] >> 16);
            }
            if (issuer) __builtin_amdgcn_s_wait_tensorcnt(0);
        }
        __syncthreads();
    }

#pragma unroll
    for (int n = 0; n < 8; ++n)
#pragma unroll
        for (int r = 0; r < 8; ++r) {
            int row = q0 + hf * 8 + r;
            QO[(size_t)((b * S_CONST + row) * NH + h) * HD + n * 16 + ln] =
                f2bf(acc[n][r] / lrow[r]);
        }
}

// ---------------------------------------------------------------------------
extern "C" void kernel_launch(void* const* d_in, const int* in_sizes, int n_in,
                              void* d_out, int out_size, void* d_ws, size_t ws_size,
                              hipStream_t stream)
{
    (void)in_sizes; (void)n_in; (void)out_size; (void)ws_size;
    const float* x  = (const float*)d_in[0];
    const float* wq = (const float*)d_in[1];
    const float* wk = (const float*)d_in[2];
    const float* wv = (const float*)d_in[3];
    const float* wo = (const float*)d_in[4];
    const float* fc = (const float*)d_in[5];
    const float* fs = (const float*)d_in[6];
    // d_in[7] = start_pos == 0

    const int    M       = B_CONST * S_CONST;        // 4096
    const int    K       = DIM_CONST;                // 4096
    const size_t qElems  = (size_t)M * DIM_CONST;    // 16M
    const size_t kvElems = (size_t)M * NKV * HD;     // 4M

    char* ws = (char*)d_ws;
    unsigned short* xb   = (unsigned short*)ws;                                 // 32 MB
    unsigned short* wtb  = (unsigned short*)(ws + qElems * 2);                  // 32 MB (wq^T, later wo^T)
    unsigned short* wktb = (unsigned short*)(ws + qElems * 4);                  // 8 MB
    unsigned short* wvtb = (unsigned short*)(ws + qElems * 4 + kvElems * 2);    // 8 MB
    unsigned short* qb   = (unsigned short*)(ws + qElems * 4 + kvElems * 4);    // 32 MB (Q, then O)
    unsigned short* kb   = (unsigned short*)(ws + qElems * 6 + kvElems * 4);    // 8 MB
    unsigned short* vb   = (unsigned short*)(ws + qElems * 6 + kvElems * 6);    // 8 MB

    f32_to_bf16<<<(int)(qElems / 256), 256, 0, stream>>>(x, xb, (int)qElems);

    transpose_conv<<<dim3(DIM_CONST / 32, K / 32), 256, 0, stream>>>(wq, wtb, K, DIM_CONST);
    gemm_tdm<true><<<dim3(DIM_CONST / 128, M / 128), 256, 0, stream>>>(
        xb, wtb, qb, M, DIM_CONST, K);

    transpose_conv<<<dim3((NKV * HD) / 32, K / 32), 256, 0, stream>>>(wk, wktb, K, NKV * HD);
    gemm_tdm<true><<<dim3((NKV * HD) / 128, M / 128), 256, 0, stream>>>(
        xb, wktb, kb, M, NKV * HD, K);

    transpose_conv<<<dim3((NKV * HD) / 32, K / 32), 256, 0, stream>>>(wv, wvtb, K, NKV * HD);
    gemm_tdm<true><<<dim3((NKV * HD) / 128, M / 128), 256, 0, stream>>>(
        xb, wvtb, vb, M, NKV * HD, K);

    {
        int totq = M * NH * (HD / 2);
        rope_inplace_bf16<<<(totq + 255) / 256, 256, 0, stream>>>(
            qb, fc, fs, NH, 0.08838834764831845f, totq);
        int totk = M * NKV * (HD / 2);
        rope_inplace_bf16<<<(totk + 255) / 256, 256, 0, stream>>>(
            kb, fc, fs, NKV, 1.0f, totk);
    }

    flash_attn<<<B_CONST * NH * (S_CONST / 64), 128, 0, stream>>>(kb, vb, qb);

    transpose_conv<<<dim3(DIM_CONST / 32, K / 32), 256, 0, stream>>>(wo, wtb, K, DIM_CONST);
    gemm_tdm<false><<<dim3(DIM_CONST / 128, M / 128), 256, 0, stream>>>(
        qb, wtb, (float*)d_out, M, DIM_CONST, K);
}
